// FlashAttentionV2_20830591385845
// MI455X (gfx1250) — matrix operand order repaired and checked by reading, not compiled, not run on hardware
//
#include <hip/hip_runtime.h>
#include <hip/hip_bf16.h>
#include <math.h>

typedef __attribute__((ext_vector_type(16))) __bf16 v16bf;
typedef __attribute__((ext_vector_type(8)))  float  v8f;

#define NTOK  8192
#define DHEAD 128
#define BM    64      // query rows per block (16 per wave)
#define BN    64      // key rows per iteration
#define NTHREADS 128  // 4 waves

// FlashAttention-2, transposed-scores formulation:
//   S^T = K . Q^T            (A = K tile [16key x 32d], B = Q^T [32d x 16q])
//   O^T = V^T . P^T          (A = V^T tile [16d x 32key], B = P^T [32key x 16q])
// C-layout of S^T puts one query's key-scores in lanes ln and ln+16 -> in-lane softmax + xor-16 merge.
__global__ __launch_bounds__(NTHREADS)
void fa2t_wmma_bf16_kernel(const float* __restrict__ Q,
                           const float* __restrict__ K,
                           const float* __restrict__ V,
                           float* __restrict__ O) {
    __shared__ __bf16 ldsK[BN * DHEAD];    // row-major  [key][d]   16KB
    __shared__ __bf16 ldsVT[DHEAD * BN];   // transposed [d][key]   16KB

    const int tid  = threadIdx.x;
    const int wave = tid >> 5;
    const int lane = tid & 31;
    const int ln   = lane & 15;
    const int kh   = lane >> 4;

    const int qrow0 = blockIdx.x * BM + wave * 16;
    // fold softmax scale and log2(e) into Q so probs come from native exp2
    const float qs = 0.08838834764831845f * 1.4426950408889634f;

    // ---- Q as B-fragments (B-layout 32x16: elem i -> d = kc*32 + kh*16 + i) ----
    v16bf qb[4];
    {
        const float* qrow = Q + (size_t)(qrow0 + ln) * DHEAD + kh * 16;
        #pragma unroll
        for (int kc = 0; kc < 4; ++kc)
            #pragma unroll
            for (int i = 0; i < 16; ++i)
                qb[kc][i] = (__bf16)(qrow[kc * 32 + i] * qs);
    }

    // O^T accumulators: 8 d-tiles of 16x16 f32 (lane = query, vgpr r -> d = dt*16+kh*8+r)
    v8f acc[8];
    #pragma unroll
    for (int dt = 0; dt < 8; ++dt)
        #pragma unroll
        for (int r = 0; r < 8; ++r) acc[dt][r] = 0.0f;

    float m = -__builtin_huge_valf();   // running max (base-2 scaled units)
    float l = 0.0f;                     // running sum

    const int njb = NTOK / BN;
    for (int j = 0; j < njb; ++j) {
        __syncthreads();  // protect LDS tiles still being read by previous P.V

        // ---- Stage K (row-major) and V (transposed) into LDS as bf16 ----
        {
            const float4* ksrc = (const float4*)(K + (size_t)j * BN * DHEAD);
            #pragma unroll
            for (int t = 0; t < (BN * DHEAD / 4) / NTHREADS; ++t) {
                const int idx  = t * NTHREADS + tid;
                const int base = idx * 4;
                float4 kv = ksrc[idx];
                ldsK[base + 0] = (__bf16)kv.x; ldsK[base + 1] = (__bf16)kv.y;
                ldsK[base + 2] = (__bf16)kv.z; ldsK[base + 3] = (__bf16)kv.w;
            }
            // V: units of 2 rows x 4 cols -> transposed pair stores (b32-mergeable)
            const float* vbase = V + (size_t)j * BN * DHEAD;
            #pragma unroll
            for (int t = 0; t < (BN * DHEAD / 8) / NTHREADS; ++t) {
                const int u  = t * NTHREADS + tid;   // 1024 units
                const int r  = (u & 31) * 2;         // key row pair
                const int c  = (u >> 5) * 4;         // d col group
                float4 a = *(const float4*)(vbase + (size_t)r * DHEAD + c);
                float4 b = *(const float4*)(vbase + (size_t)(r + 1) * DHEAD + c);
                ldsVT[(c + 0) * BN + r] = (__bf16)a.x; ldsVT[(c + 0) * BN + r + 1] = (__bf16)b.x;
                ldsVT[(c + 1) * BN + r] = (__bf16)a.y; ldsVT[(c + 1) * BN + r + 1] = (__bf16)b.y;
                ldsVT[(c + 2) * BN + r] = (__bf16)a.z; ldsVT[(c + 2) * BN + r + 1] = (__bf16)b.z;
                ldsVT[(c + 3) * BN + r] = (__bf16)a.w; ldsVT[(c + 3) * BN + r + 1] = (__bf16)b.w;
            }
            if (j + 1 < njb) {
                __builtin_prefetch(K + (size_t)(j + 1) * BN * DHEAD + tid * 64, 0, 1);
                __builtin_prefetch(V + (size_t)(j + 1) * BN * DHEAD + tid * 64, 0, 1);
            }
        }
        __syncthreads();

        // ---- S^T = K . Q^T  (4 key sub-tiles x 4 d-chunks) ----
        v8f st[4];
        #pragma unroll
        for (int nt = 0; nt < 4; ++nt) {
            #pragma unroll
            for (int r = 0; r < 8; ++r) st[nt][r] = 0.0f;
            #pragma unroll
            for (int kc = 0; kc < 4; ++kc) {
                // A-layout 16x32: lane ln = key row; elems = 16 contiguous d (d = kc*32 + kh*16 + i)
                v16bf ka;
                const __bf16* kr = &ldsK[(nt * 16 + ln) * DHEAD + kc * 32 + kh * 16];
                #pragma unroll
                for (int i = 0; i < 8; ++i) { ka[i] = kr[i]; ka[8 + i] = kr[8 + i]; }
                st[nt] = __builtin_amdgcn_wmma_f32_16x16x32_bf16(
                    false, ka, false, qb[kc], (short)0, st[nt], false, false);
            }
        }

        // ---- Online softmax: 32 of query `ln`'s 64 scores live in this lane, the other 32 in lane ^ 16 ----
        float mx = st[0][0];
        #pragma unroll
        for (int nt = 0; nt < 4; ++nt)
            #pragma unroll
            for (int r = 0; r < 8; ++r) mx = fmaxf(mx, st[nt][r]);
        mx = fmaxf(mx, __shfl_xor(mx, 16, 32));   // merge the two key-halves (same query)
        const float mnew = fmaxf(m, mx);

        float rs = 0.0f;
        #pragma unroll
        for (int nt = 0; nt < 4; ++nt)
            #pragma unroll
            for (int r = 0; r < 8; ++r) {
                float p = exp2f(st[nt][r] - mnew);
                st[nt][r] = p;
                rs += p;
            }
        rs += __shfl_xor(rs, 16, 32);

        const float alpha = exp2f(m - mnew);      // exp2(-inf)=0 on first block
        l = l * alpha + rs;
        m = mnew;
        #pragma unroll
        for (int dt = 0; dt < 8; ++dt)
            #pragma unroll
            for (int r = 0; r < 8; ++r) acc[dt][r] *= alpha;

        // ---- O^T += V^T . P^T : build P^T B-fragments with one xor-16 exchange ----
        #pragma unroll
        for (int kc2 = 0; kc2 < 2; ++kc2) {
            v16bf pb;   // B-layout: elem i -> key = kc2*32 + kh*16 + i, col = query ln
            #pragma unroll
            for (int r = 0; r < 8; ++r) {
                const float own  = kh ? st[2 * kc2 + 1][r] : st[2 * kc2][r];
                const float send = kh ? st[2 * kc2][r]     : st[2 * kc2 + 1][r];
                const float recv = __shfl_xor(send, 16, 32);
                const float lo = kh ? recv : own;
                const float hi = kh ? own  : recv;
                pb[r]     = (__bf16)lo;
                pb[8 + r] = (__bf16)hi;
            }
            #pragma unroll
            for (int dt = 0; dt < 8; ++dt) {
                // A-layout of V^T: lane ln = d row dt*16+ln; 16 contiguous keys (key = kc2*32 + kh*16 + i)
                v16bf va;
                const __bf16* vr = &ldsVT[(dt * 16 + ln) * BN + kc2 * 32 + kh * 16];
                #pragma unroll
                for (int i = 0; i < 8; ++i) { va[i] = vr[i]; va[8 + i] = vr[8 + i]; }
                acc[dt] = __builtin_amdgcn_wmma_f32_16x16x32_bf16(
                    false, va, false, pb, (short)0, acc[dt], false, false);
            }
        }
    }

    // ---- Epilogue: O[q][d] = acc / l  (8 contiguous floats per d-tile) ----
    const float rl = 1.0f / l;
    #pragma unroll
    for (int dt = 0; dt < 8; ++dt) {
        float* op = O + (size_t)(qrow0 + ln) * DHEAD + dt * 16 + kh * 8;
        #pragma unroll
        for (int r = 0; r < 8; ++r) op[r] = acc[dt][r] * rl;
    }
}

extern "C" void kernel_launch(void* const* d_in, const int* in_sizes, int n_in,
                              void* d_out, int out_size, void* d_ws, size_t ws_size,
                              hipStream_t stream) {
    (void)in_sizes; (void)n_in; (void)out_size; (void)d_ws; (void)ws_size;
    const float* Q = (const float*)d_in[0];
    const float* K = (const float*)d_in[1];
    const float* V = (const float*)d_in[2];
    float* O = (float*)d_out;
    dim3 grid(NTOK / BM);
    dim3 block(NTHREADS);
    hipLaunchKernelGGL(fa2t_wmma_bf16_kernel, grid, block, 0, stream, Q, K, V, O);
}
